// MHA_28329604284698
// MI455X (gfx1250) — compile-verified, matrix operand order checked by reading, not run on hardware
//
#include <hip/hip_runtime.h>

// ---------------------------------------------------------------------------
// Types for CDNA5 WMMA (wave32): V_WMMA_F32_16X16X32_BF16
// ---------------------------------------------------------------------------
typedef __attribute__((ext_vector_type(16))) __bf16 bf16x16;
typedef __attribute__((ext_vector_type(8)))  __bf16 bf16x8;
typedef __attribute__((ext_vector_type(8)))  float  f32x8;
typedef __attribute__((ext_vector_type(4)))  unsigned u32x4;
typedef __attribute__((ext_vector_type(8)))  int      i32x8;
typedef __attribute__((ext_vector_type(4)))  int      i32x4;

__device__ __forceinline__ f32x8 wmma_bf16(bf16x16 a, bf16x16 b, f32x8 c) {
  // (neg_a, A, neg_b, B, c_mod, C, reuse_a, reuse_b)
  return __builtin_amdgcn_wmma_f32_16x16x32_bf16(false, a, false, b,
                                                 (short)0, c, false, false);
}

__device__ __forceinline__ bf16x16 join8(bf16x8 lo, bf16x8 hi) {
  return __builtin_shufflevector(lo, hi, 0, 1, 2, 3, 4, 5, 6, 7,
                                 8, 9, 10, 11, 12, 13, 14, 15);
}

__device__ __forceinline__ f32x8 zero8() {
  f32x8 z;
#pragma unroll
  for (int i = 0; i < 8; ++i) z[i] = 0.0f;
  return z;
}

// Problem constants
#define BB 2
#define SS 2048
#define DD 2048
#define HH 16
#define HD 128
#define HALF 64
#define MTOT (BB * SS)   // 4096

// ---------------------------------------------------------------------------
// TDM: async 2D tile load Global->LDS (Tensor Data Mover, gfx1250).
// Loads a [rows=64][cols=128] bf16 tile whose rows are contiguous 256B in
// memory (row stride 128 elems), and pads the LDS destination by 4 DWORDs
// every 64 DWORDs, producing exactly the 136-element LDS row stride used by
// the WMMA fragment readers (pad_interval=5 -> 64 DW, pad_amount=3 -> 4 DW).
// D# layout per CDNA5 ISA.
// ---------------------------------------------------------------------------
__device__ __forceinline__ void tdm_load_k_tile(unsigned lds_addr,
                                                unsigned long long gaddr) {
  u32x4 g0;
  g0[0] = 1u;                                     // count=1, user mode
  g0[1] = lds_addr;                               // LDS byte address
  g0[2] = (unsigned)(gaddr & 0xffffffffull);      // global_addr[31:0]
  g0[3] = (unsigned)((gaddr >> 32) & 0x01ffffffu) // global_addr[56:32]
          | (2u << 30);                           // type=2 ("image")

  i32x8 g1;
  g1[0] = (int)((1u << 16)        // data_size = 2 bytes
                | (1u << 20)      // pad_enable
                | (5u << 22)      // pad_interval: 64 DWORDs
                | (3u << 25));    // pad_amount: 4 DWORDs
  g1[1] = (int)(128u << 16);      // tensor_dim0[15:0]=128 in bits [31:16]
  g1[2] = (int)(2048u << 16);     // tensor_dim0[31:16]=0 | tensor_dim1[15:0]=2048
  g1[3] = (int)(128u << 16);      // tensor_dim1[31:16]=0 | tile_dim0=128
  g1[4] = 64;                     // tile_dim1=64, tile_dim2=0
  g1[5] = 128;                    // tensor_dim0_stride[31:0]=128
  g1[6] = 0;                      // dim0_stride[47:32]=0, dim1_stride[15:0]=0
  g1[7] = 0;                      // dim1_stride[47:16]=0

  i32x4 gz = {0, 0, 0, 0};
#if __has_include(<hip/amd_detail/amd_gfx1250_TDM.h>)
  i32x8 gz8 = {0, 0, 0, 0, 0, 0, 0, 0};
  __builtin_amdgcn_tensor_load_to_lds(g0, g1, gz, gz, gz8, 0);
#else
  __builtin_amdgcn_tensor_load_to_lds(g0, g1, gz, gz, 0);
#endif
}

// ---------------------------------------------------------------------------
// Generic GEMM: C[M,N] = A[M,K] (f32, row-major) * B[K,N] (f32, row-major)
// bf16 WMMA with f32 accumulation. Tile 128x128, BK=32, 256 threads = 8 waves.
// Wave grid 4(m) x 2(n); each wave computes 32x64 via 2x4 16x16 WMMA tiles.
// ---------------------------------------------------------------------------
template <typename OutT>
__global__ __launch_bounds__(256) void gemm_bf16_wmma(
    const float* __restrict__ A, const float* __restrict__ B,
    OutT* __restrict__ C, int M, int N, int K) {
  constexpr int LD = 40;  // bf16 elems per LDS row (80B, 16B-aligned stride)
  __shared__ __align__(16) __bf16 sA[128 * LD];  // [row][k]
  __shared__ __align__(16) __bf16 sB[128 * LD];  // transposed: [n][k]

  const int t    = threadIdx.x;
  const int lane = t & 31;
  const int wave = t >> 5;
  const int wm   = wave >> 1;  // 0..3
  const int wn   = wave & 1;   // 0..1
  const int ln   = lane & 15;
  const int sel  = lane >> 4;  // which half-wave (K-offset select)

  const int tileM = blockIdx.y * 128;
  const int tileN = blockIdx.x * 128;

  f32x8 acc[2][4];
#pragma unroll
  for (int i = 0; i < 2; ++i)
#pragma unroll
    for (int j = 0; j < 4; ++j) acc[i][j] = zero8();

  for (int kb = 0; kb < K; kb += 32) {
    // ---- stage A tile (128x32 f32 -> bf16) ----
#pragma unroll
    for (int i = 0; i < 4; ++i) {
      int idx = t + i * 256;           // 0..1023
      int row = idx >> 3;              // 0..127
      int c4  = (idx & 7) << 2;        // 0,4,..,28
      float4 v = *(const float4*)(A + (size_t)(tileM + row) * K + kb + c4);
      __bf16* d = sA + row * LD + c4;
      d[0] = (__bf16)v.x; d[1] = (__bf16)v.y;
      d[2] = (__bf16)v.z; d[3] = (__bf16)v.w;
    }
    // ---- stage B tile transposed (32x128 f32 -> bf16 [n][k]) ----
#pragma unroll
    for (int i = 0; i < 4; ++i) {
      int idx = t + i * 256;
      int kr  = idx >> 5;              // 0..31
      int n4  = (idx & 31) << 2;       // 0..124
      float4 v = *(const float4*)(B + (size_t)(kb + kr) * N + tileN + n4);
      sB[(n4 + 0) * LD + kr] = (__bf16)v.x;
      sB[(n4 + 1) * LD + kr] = (__bf16)v.y;
      sB[(n4 + 2) * LD + kr] = (__bf16)v.z;
      sB[(n4 + 3) * LD + kr] = (__bf16)v.w;
    }
    if (kb + 32 < K) {  // hint next tiles into cache
      __builtin_prefetch(A + (size_t)(tileM + (t >> 1)) * K + kb + 32, 0, 1);
      __builtin_prefetch(B + (size_t)(kb + 32 + (t >> 3)) * N + tileN, 0, 1);
    }
    __syncthreads();

    // ---- fragments & 8 WMMAs ----
    bf16x16 af[2], bfr[4];
#pragma unroll
    for (int i = 0; i < 2; ++i) {
      const __bf16* p = sA + (wm * 32 + i * 16 + ln) * LD + sel * 8;
      af[i] = join8(*(const bf16x8*)p, *(const bf16x8*)(p + 16));
    }
#pragma unroll
    for (int j = 0; j < 4; ++j) {
      const __bf16* p = sB + (wn * 64 + j * 16 + ln) * LD + sel * 8;
      bfr[j] = join8(*(const bf16x8*)p, *(const bf16x8*)(p + 16));
    }
#pragma unroll
    for (int i = 0; i < 2; ++i)
#pragma unroll
      for (int j = 0; j < 4; ++j) acc[i][j] = wmma_bf16(af[i], bfr[j], acc[i][j]);
    __syncthreads();
  }

  // ---- epilogue: C/D layout -> lanes 0-15: M=r, lanes 16-31: M=r+8 ----
#pragma unroll
  for (int i = 0; i < 2; ++i) {
    int m0 = tileM + wm * 32 + i * 16 + sel * 8;
#pragma unroll
    for (int j = 0; j < 4; ++j) {
      int n = tileN + wn * 64 + j * 16 + ln;
#pragma unroll
      for (int r = 0; r < 8; ++r)
        C[(size_t)(m0 + r) * N + n] = (OutT)acc[i][j][r];
    }
  }
}

// ---------------------------------------------------------------------------
// RoPE + split + layout change: qkv bf16 [4096][6144] ->
//   Q,K (rotated; Q pre-scaled by 1/sqrt(hd)), V  as bf16 [B,H,S,hd]
// ---------------------------------------------------------------------------
__global__ __launch_bounds__(256) void rope_split(
    const __bf16* __restrict__ qkv, __bf16* __restrict__ Qb,
    __bf16* __restrict__ Kb, __bf16* __restrict__ Vb) {
  int idx = blockIdx.x * blockDim.x + threadIdx.x;  // B*S*H*half = 4194304
  if (idx >= BB * SS * HH * HALF) return;
  int i = idx & 63;
  int h = (idx >> 6) & 15;
  int s = (idx >> 10) & 2047;
  int b = idx >> 21;

  const size_t row = (size_t)(b * SS + s) * (3 * DD);
  const int col = h * HD + 2 * i;
  float qe = (float)qkv[row + col],            qo = (float)qkv[row + col + 1];
  float ke = (float)qkv[row + DD + col],       ko = (float)qkv[row + DD + col + 1];
  float ve = (float)qkv[row + 2 * DD + col],   vo = (float)qkv[row + 2 * DD + col + 1];

  float inv = __powf(10000.0f, -(2.0f * (float)i) / (float)HD);
  float ang = (float)s * inv;
  float sn, cs;
  __sincosf(ang, &sn, &cs);
  const float scale = 0.08838834764831845f;  // 1/sqrt(128)

  float qre = (qe * cs - qo * sn) * scale;
  float qim = (qe * sn + qo * cs) * scale;
  float kre = ke * cs - ko * sn;
  float kim = ke * sn + ko * cs;

  size_t d = ((size_t)(b * HH + h) * SS + s) * HD + 2 * i;
  Qb[d] = (__bf16)qre; Qb[d + 1] = (__bf16)qim;
  Kb[d] = (__bf16)kre; Kb[d + 1] = (__bf16)kim;
  Vb[d] = (__bf16)ve;  Vb[d + 1] = (__bf16)vo;
}

// ---------------------------------------------------------------------------
// Flash attention (causal): grid = (S/128, B*H), 256 threads = 8 waves.
// Each wave owns 16 query rows. K-tile (64x128) is DMA'd into LDS by the TDM
// (wave 0 issues, pads to the 136-elem stride); V^T tile is staged manually
// (transpose). P goes through a private per-wave LDS slab to become an A-frag.
// ---------------------------------------------------------------------------
__global__ __launch_bounds__(256) void flash_attn(
    const __bf16* __restrict__ Qb, const __bf16* __restrict__ Kb,
    const __bf16* __restrict__ Vb, float* __restrict__ O) {
  constexpr int LDK = 136;  // 272B stride, 16B aligned (TDM pad reproduces it)
  constexpr int LDV = 72;   // 144B stride, 16B aligned
  __shared__ __align__(16) __bf16 sK[64 * LDK];    // [key][hd]
  __shared__ __align__(16) __bf16 sVt[HD * LDV];   // [hd][key]
  __shared__ __align__(16) __bf16 sP[8][16 * 64];  // per-wave P tile

  const int t    = threadIdx.x;
  const int lane = t & 31;
  const int wave = t >> 5;
  const int ln   = lane & 15;
  const int sel  = lane >> 4;

  const int bh    = blockIdx.y;            // b*H + h
  const int qtile = blockIdx.x;            // 0..15
  const int qrow  = qtile * 128 + wave * 16;
  const size_t base = (size_t)bh * SS * HD;

  // LDS byte offset of sK (generic LDS pointer: addr[31:0] is the LDS offset)
  const unsigned sK_lds = (unsigned)(unsigned long long)(const void*)sK;

  // Q fragments for this wave's 16 rows (hd = 4 k-chunks of 32)
  bf16x16 qf[4];
  {
    const __bf16* qp = Qb + base + (size_t)(qrow + ln) * HD;
#pragma unroll
    for (int kc = 0; kc < 4; ++kc) {
      const __bf16* p = qp + kc * 32 + sel * 8;
      qf[kc] = join8(*(const bf16x8*)p, *(const bf16x8*)(p + 16));
    }
  }

  f32x8 oacc[8];
#pragma unroll
  for (int j = 0; j < 8; ++j) oacc[j] = zero8();
  float mrow[8], lrow[8];
#pragma unroll
  for (int r = 0; r < 8; ++r) { mrow[r] = -1.0e30f; lrow[r] = 0.0f; }

  const int kmax = qtile * 128 + 127;
  for (int kb = 0; kb <= kmax; kb += 64) {
    // ---- K tile [64][hd] via Tensor Data Mover (wave 0 issues the DMA) ----
    if (wave == 0) {
      unsigned long long gK =
          (unsigned long long)(Kb + base + (size_t)kb * HD);
      tdm_load_k_tile(sK_lds, gK);
    }
    // ---- stage V tile transposed [hd][key] (overlaps with the TDM) ----
#pragma unroll
    for (int i = 0; i < 16; ++i) {
      int idx = t + i * 256;  // 0..4095
      int key = idx >> 6;
      int c   = idx & 63;     // pair of hd elems
      unsigned v = *(const unsigned*)(Vb + base + (size_t)(kb + key) * HD + c * 2);
      unsigned short l0 = (unsigned short)(v & 0xffffu);
      unsigned short h0 = (unsigned short)(v >> 16);
      sVt[(2 * c + 0) * LDV + key] = __builtin_bit_cast(__bf16, l0);
      sVt[(2 * c + 1) * LDV + key] = __builtin_bit_cast(__bf16, h0);
    }
    if (kb + 64 <= kmax)
      __builtin_prefetch(Vb + base + (size_t)(kb + 64 + (t >> 2)) * HD, 0, 1);
    if (wave == 0) __builtin_amdgcn_s_wait_tensorcnt(0);
    __syncthreads();

    // ---- scores: 4 key-subtiles x 4 hd-chunks = 16 WMMA ----
    f32x8 sc[4];
#pragma unroll
    for (int j = 0; j < 4; ++j) {
      sc[j] = zero8();
#pragma unroll
      for (int kc = 0; kc < 4; ++kc) {
        const __bf16* p = sK + (j * 16 + ln) * LDK + kc * 32 + sel * 8;
        bf16x16 kf = join8(*(const bf16x8*)p, *(const bf16x8*)(p + 16));
        sc[j] = wmma_bf16(qf[kc], kf, sc[j]);
      }
    }

    // ---- causal mask (lane = key col, VGPR r = query row) ----
    if (kb + 63 > qrow) {
#pragma unroll
      for (int j = 0; j < 4; ++j) {
        int n = kb + j * 16 + ln;
#pragma unroll
        for (int r = 0; r < 8; ++r) {
          int m = qrow + sel * 8 + r;
          if (n > m) sc[j][r] = -1.0e30f;
        }
      }
    }

    // ---- online softmax, per row (reduce across 16-lane half-waves) ----
    __bf16* sPw = sP[wave];
#pragma unroll
    for (int r = 0; r < 8; ++r) {
      float v = fmaxf(fmaxf(sc[0][r], sc[1][r]), fmaxf(sc[2][r], sc[3][r]));
#pragma unroll
      for (int mk = 1; mk <= 8; mk <<= 1) v = fmaxf(v, __shfl_xor(v, mk, 32));
      float mn   = fmaxf(mrow[r], v);
      float corr = __expf(mrow[r] - mn);
      mrow[r] = mn;
      float rs = 0.0f;
#pragma unroll
      for (int j = 0; j < 4; ++j) {
        float p = __expf(sc[j][r] - mn);
        sc[j][r] = p;
        rs += p;
      }
#pragma unroll
      for (int mk = 1; mk <= 8; mk <<= 1) rs += __shfl_xor(rs, mk, 32);
      lrow[r] = lrow[r] * corr + rs;
#pragma unroll
      for (int j2 = 0; j2 < 8; ++j2) oacc[j2][r] *= corr;
      // stash P row into per-wave LDS (C-layout -> memory)
#pragma unroll
      for (int j = 0; j < 4; ++j)
        sPw[(sel * 8 + r) * 64 + j * 16 + ln] = (__bf16)sc[j][r];
    }

    // ---- O += P(16x64) * V(64x128): 2 k-chunks x 8 hd-tiles ----
    bf16x16 pf[2];
#pragma unroll
    for (int kk = 0; kk < 2; ++kk) {
      const __bf16* p = sPw + ln * 64 + kk * 32 + sel * 8;
      pf[kk] = join8(*(const bf16x8*)p, *(const bf16x8*)(p + 16));
    }
#pragma unroll
    for (int j2 = 0; j2 < 8; ++j2) {
#pragma unroll
      for (int kk = 0; kk < 2; ++kk) {
        const __bf16* p = sVt + (j2 * 16 + ln) * LDV + kk * 32 + sel * 8;
        bf16x16 vf = join8(*(const bf16x8*)p, *(const bf16x8*)(p + 16));
        oacc[j2] = wmma_bf16(pf[kk], vf, oacc[j2]);
      }
    }
    __syncthreads();
  }

  // ---- epilogue: O[b, s, h*hd + d] fp32, normalized by row sums ----
  const int b = bh >> 4;
  const int h = bh & 15;
#pragma unroll
  for (int j2 = 0; j2 < 8; ++j2) {
    int d = h * HD + j2 * 16 + ln;
#pragma unroll
    for (int r = 0; r < 8; ++r) {
      int srow = qrow + sel * 8 + r;
      O[((size_t)(b * SS + srow)) * DD + d] = oacc[j2][r] / lrow[r];
    }
  }
}

// ---------------------------------------------------------------------------
// Host-side launcher
// ---------------------------------------------------------------------------
extern "C" void kernel_launch(void* const* d_in, const int* in_sizes, int n_in,
                              void* d_out, int out_size, void* d_ws,
                              size_t ws_size, hipStream_t stream) {
  const float* x    = (const float*)d_in[0];  // [2,2048,2048]
  const float* Wqkv = (const float*)d_in[1];  // [2048,6144]
  const float* Wo   = (const float*)d_in[2];  // [2048,2048]
  float* out = (float*)d_out;                 // [2,2048,2048]

  // workspace layout
  __bf16* qkv = (__bf16*)d_ws;                          // 4096*6144 bf16
  __bf16* Qb  = qkv + (size_t)MTOT * 3 * DD;            // 4096*2048 bf16
  __bf16* Kb  = Qb + (size_t)MTOT * DD;
  __bf16* Vb  = Kb + (size_t)MTOT * DD;
  float*  Obf = (float*)(Vb + (size_t)MTOT * DD);       // 4096*2048 f32

  // 1) qkv = x @ Wqkv  (M=4096, N=6144, K=2048) -> bf16
  gemm_bf16_wmma<__bf16><<<dim3(6144 / 128, MTOT / 128), 256, 0, stream>>>(
      x, Wqkv, qkv, MTOT, 3 * DD, DD);

  // 2) RoPE + split into Q,K,V [B,H,S,hd] bf16 (Q pre-scaled)
  rope_split<<<(BB * SS * HH * HALF) / 256, 256, 0, stream>>>(qkv, Qb, Kb, Vb);

  // 3) causal flash attention -> O [B,S,D] f32
  flash_attn<<<dim3(SS / 128, BB * HH), 256, 0, stream>>>(Qb, Kb, Vb, Obf);

  // 4) out = O @ Wo  (M=4096, N=2048, K=2048) -> f32
  gemm_bf16_wmma<float><<<dim3(DD / 128, MTOT / 128), 256, 0, stream>>>(
      Obf, Wo, out, MTOT, DD, DD);
}
